// Adapter_39522289058457
// MI455X (gfx1250) — matrix operand order repaired and checked by reading, not compiled, not run on hardware
//
#include <hip/hip_runtime.h>
#include <hip/hip_bf16.h>
#include <math.h>

typedef __attribute__((ext_vector_type(16))) _Float16 v16h;
typedef __attribute__((ext_vector_type(8)))  _Float16 v8h;
typedef __attribute__((ext_vector_type(4)))  _Float16 v4h;
typedef __attribute__((ext_vector_type(8)))  float    v8f;

#define D_DIM 2048
#define A_DIM 256
#define TM    32                  // tokens per block (2 WMMA M-tiles)
#define XS    (D_DIM + 8)         // padded LDS row stride (f16) for xbuf
#define HS    (A_DIM + 8)         // padded LDS row stride (f16) for hbuf

// ---------------------------------------------------------------------------
// Weight swizzle: fp32 row-major W[k][n] -> f16 WMMA B-fragment-major layout.
// Fragment = 32 lanes x 16 f16. For v_wmma_f32_16x16x32_f16 the B matrix
// (32x16, KxN) lane layout is: n = lane&15, khalf = lane>>4, element j holds
// K = (j>>3)*16 + khalf*8 + (j&7). Fragment index for w_down: f = nt*64 + kc (N=256 -> 16 nt,
// K=2048 -> 64 kc). For w_up: f = nt*8 + kc (N=2048 -> 128 nt, K=256 -> 8 kc).
// ---------------------------------------------------------------------------
__global__ __launch_bounds__(256) void swizzle_wd(const float* __restrict__ w,
                                                  _Float16* __restrict__ o) {
  int idx = blockIdx.x * 256 + threadIdx.x;          // 2048*256 = 524288 elems
  int f = idx >> 9, l = (idx >> 4) & 31, j = idx & 15;
  int nt = f >> 6, kc = f & 63;
  int n = nt * 16 + (l & 15);
  int k = kc * 32 + (j >> 3) * 16 + (l >> 4) * 8 + (j & 7);
  o[idx] = (_Float16)w[k * A_DIM + n];
}

__global__ __launch_bounds__(256) void swizzle_wu(const float* __restrict__ w,
                                                  _Float16* __restrict__ o) {
  int idx = blockIdx.x * 256 + threadIdx.x;          // 256*2048 = 524288 elems
  int f = idx >> 9, l = (idx >> 4) & 31, j = idx & 15;
  int nt = f >> 3, kc = f & 7;
  int n = nt * 16 + (l & 15);
  int k = kc * 32 + (j >> 3) * 16 + (l >> 4) * 8 + (j & 7);
  o[idx] = (_Float16)w[k * D_DIM + n];
}

// ---------------------------------------------------------------------------
// Fused adapter: LayerNorm -> down-proj(WMMA) -> GELU -> up-proj(WMMA) -> +res
// Block: 256 threads = 8 wave32, 32 tokens.
// ---------------------------------------------------------------------------
__global__ __launch_bounds__(256) void adapter_kernel(
    const float* __restrict__ hs, const float* __restrict__ gamma,
    const float* __restrict__ beta, const float* __restrict__ b_down,
    const float* __restrict__ b_up, const _Float16* __restrict__ wsD,
    const _Float16* __restrict__ wsU, float* __restrict__ out) {
  extern __shared__ char smem_raw[];
  _Float16* xbuf = (_Float16*)smem_raw;                                   // [TM][XS]
  _Float16* hbuf = (_Float16*)(smem_raw + (size_t)TM * XS * sizeof(_Float16)); // [TM][HS]

  const int tid  = threadIdx.x;
  const int wave = tid >> 5;
  const int lane = tid & 31;
  const int m16  = lane & 15;       // row within 16-row tile (A), col within tile (B/C)
  const int kh   = lane >> 4;       // K-half selector per ISA fragment layout
  const int tok0 = blockIdx.x * TM;

  // ---------------- Phase 1: LayerNorm (wave handles 4 tokens) -------------
  for (int c = 0; c < 4; ++c) {
    const int row = wave * 4 + c;
    const float4* xr = (const float4*)(hs + (size_t)(tok0 + row) * D_DIM);
    float4 xv[16];
    float s = 0.f, q = 0.f;
#pragma unroll
    for (int i = 0; i < 16; ++i) {
      xv[i] = xr[lane + 32 * i];
      s += xv[i].x + xv[i].y + xv[i].z + xv[i].w;
      q += xv[i].x * xv[i].x + xv[i].y * xv[i].y + xv[i].z * xv[i].z + xv[i].w * xv[i].w;
    }
#pragma unroll
    for (int off = 16; off > 0; off >>= 1) {
      s += __shfl_xor(s, off, 32);
      q += __shfl_xor(q, off, 32);
    }
    const float mu  = s * (1.0f / D_DIM);
    const float var = q * (1.0f / D_DIM) - mu * mu;
    const float rs  = rsqrtf(var + 1e-5f);
#pragma unroll
    for (int i = 0; i < 16; ++i) {
      const int j4 = lane + 32 * i;
      const float4 g = ((const float4*)gamma)[j4];
      const float4 b = ((const float4*)beta)[j4];
      v4h p;
      p[0] = (_Float16)((xv[i].x - mu) * rs * g.x + b.x);
      p[1] = (_Float16)((xv[i].y - mu) * rs * g.y + b.y);
      p[2] = (_Float16)((xv[i].z - mu) * rs * g.z + b.z);
      p[3] = (_Float16)((xv[i].w - mu) * rs * g.w + b.w);
      *(v4h*)&xbuf[row * XS + j4 * 4] = p;
    }
  }
  __syncthreads();

  // ---------------- Phase 2: down-proj (16x16x32 f16 WMMA) + GELU ----------
  // Wave owns N-tiles {2w, 2w+1}; both M-tiles; K = 2048 in 64 chunks.
  v8f acc[2][2] = {};
  for (int kc = 0; kc < 64; ++kc) {
    v16h a[2];
#pragma unroll
    for (int mt = 0; mt < 2; ++mt) {
      // 16-bit A 16x32 layout: lane m16/kh -> K runs {kh*8+0..7, 16+kh*8+0..7}
      const _Float16* p = &xbuf[(mt * 16 + m16) * XS + kc * 32 + kh * 8];
      v8h lo = *(const v8h*)p;
      v8h hi = *(const v8h*)(p + 16);
      a[mt] = __builtin_shufflevector(lo, hi, 0, 1, 2, 3, 4, 5, 6, 7,
                                      8, 9, 10, 11, 12, 13, 14, 15);
    }
#pragma unroll
    for (int s2 = 0; s2 < 2; ++s2) {
      const int ntg = wave * 2 + s2;
      v16h b = *(const v16h*)&wsD[(size_t)(ntg * 64 + kc) * 512 + lane * 16];
#pragma unroll
      for (int mt = 0; mt < 2; ++mt)
        acc[s2][mt] = __builtin_amdgcn_wmma_f32_16x16x32_f16(
            false, a[mt], false, b, (short)0, acc[s2][mt], false, false);
    }
  }
  // bias + exact GELU, scatter f32 accum (col = m16, rows r + kh*8) to LDS f16
#pragma unroll
  for (int s2 = 0; s2 < 2; ++s2) {
    const int n = (wave * 2 + s2) * 16 + m16;
    const float bias = b_down[n];
#pragma unroll
    for (int mt = 0; mt < 2; ++mt)
#pragma unroll
      for (int r = 0; r < 8; ++r) {
        const float hv = acc[s2][mt][r] + bias;
        const float ge = 0.5f * hv * (1.0f + erff(hv * 0.70710678118654752f));
        hbuf[(mt * 16 + r + kh * 8) * HS + n] = (_Float16)ge;
      }
  }
  __syncthreads();

  // ---------------- Phase 3: up-proj (WMMA) + bias + residual --------------
  // Wave owns 256 output columns [wave*256, wave*256+256); K = 256 (8 chunks).
  v16h ah[2][8];
#pragma unroll
  for (int mt = 0; mt < 2; ++mt)
#pragma unroll
    for (int kc = 0; kc < 8; ++kc) {
      const _Float16* p = &hbuf[(mt * 16 + m16) * HS + kc * 32 + kh * 8];
      v8h lo = *(const v8h*)p;
      v8h hi = *(const v8h*)(p + 16);
      ah[mt][kc] = __builtin_shufflevector(lo, hi, 0, 1, 2, 3, 4, 5, 6, 7,
                                           8, 9, 10, 11, 12, 13, 14, 15);
    }

  for (int nt = 0; nt < 16; ++nt) {
    const int d = wave * 256 + nt * 16 + m16;
    const float bias = b_up[d];
    v8f acc2[2] = {};
#pragma unroll
    for (int kc = 0; kc < 8; ++kc) {
      v16h b = *(const v16h*)&wsU[(size_t)(((wave * 16 + nt) * 8) + kc) * 512 + lane * 16];
#pragma unroll
      for (int mt = 0; mt < 2; ++mt)
        acc2[mt] = __builtin_amdgcn_wmma_f32_16x16x32_f16(
            false, ah[mt][kc], false, b, (short)0, acc2[mt], false, false);
    }
#pragma unroll
    for (int mt = 0; mt < 2; ++mt)
#pragma unroll
      for (int r = 0; r < 8; ++r) {
        const size_t gi = (size_t)(tok0 + mt * 16 + r + kh * 8) * D_DIM + d;
        out[gi] = acc2[mt][r] + bias + hs[gi];
      }
  }
}

extern "C" void kernel_launch(void* const* d_in, const int* in_sizes, int n_in,
                              void* d_out, int out_size, void* d_ws, size_t ws_size,
                              hipStream_t stream) {
  const float* hs     = (const float*)d_in[0];
  const float* gamma  = (const float*)d_in[1];
  const float* beta   = (const float*)d_in[2];
  const float* w_down = (const float*)d_in[3];
  const float* b_down = (const float*)d_in[4];
  const float* w_up   = (const float*)d_in[5];
  const float* b_up   = (const float*)d_in[6];
  float* out = (float*)d_out;

  const int tokens = in_sizes[0] / D_DIM;   // 32768

  _Float16* wsD = (_Float16*)d_ws;                  // 2048*256 f16 = 1 MB
  _Float16* wsU = wsD + (size_t)D_DIM * A_DIM;      // 256*2048 f16 = 1 MB

  // One-shot weight fp32 -> f16 fragment-major swizzle (weights live in L2).
  swizzle_wd<<<(D_DIM * A_DIM) / 256, 256, 0, stream>>>(w_down, wsD);
  swizzle_wu<<<(A_DIM * D_DIM) / 256, 256, 0, stream>>>(w_up, wsU);

  const size_t smem = (size_t)TM * XS * sizeof(_Float16) +
                      (size_t)TM * HS * sizeof(_Float16);   // 148480 B
  hipFuncSetAttribute((const void*)adapter_kernel,
                      hipFuncAttributeMaxDynamicSharedMemorySize, (int)smem);
  adapter_kernel<<<tokens / TM, 256, smem, stream>>>(hs, gamma, beta, b_down,
                                                     b_up, wsD, wsU, out);
}
